// DRMTransformerBlock_38878043963545
// MI455X (gfx1250) — compile-verified, matrix operand order checked by reading, not run on hardware
//
#include <hip/hip_runtime.h>
#include <cstdint>
#include <cstddef>

typedef __bf16 bf16;
typedef __attribute__((ext_vector_type(16))) __bf16    v16bf;
typedef __attribute__((ext_vector_type(8)))  float     v8f;
typedef __attribute__((ext_vector_type(4)))  uint32_t  v4u;
typedef __attribute__((ext_vector_type(8)))  uint32_t  v8u;

// ---------- bf16 helpers (bit-exact RNE, no reliance on __bf16 conv lowering) ----------
__device__ __forceinline__ bf16 f2bf(float f) {
    uint32_t u = __float_as_uint(f);
    uint32_t r = u + 0x7FFFu + ((u >> 16) & 1u);
    uint16_t h = (uint16_t)(r >> 16);
    return __builtin_bit_cast(bf16, h);
}
__device__ __forceinline__ float bf2f(bf16 x) {
    uint16_t h = __builtin_bit_cast(uint16_t, x);
    return __uint_as_float(((uint32_t)h) << 16);
}

// =====================================================================================
// TDM issue: DMA a [128 rows x 16 dwords] tile (row stride Kd dwords in memory) into
// LDS with 4-dword padding after every 16 dwords (-> 20-dword LDS row stride, which is
// bank-conflict free for the WMMA fragment reads). One issue covers the whole tile.
// D# layout per CDNA5 ISA (group0 128b + group1 256b).
// =====================================================================================
__device__ __forceinline__ void tdm_load_tile(uint32_t lds_addr, const void* gp,
                                              uint32_t Kd /*row len+stride, dwords*/) {
    uint64_t ga = (uint64_t)(uintptr_t)gp;
    union { uint32_t a[4]; v4u v; } g0;
    g0.a[0] = 1u;                                   // count=1, is_restore=0, gather=0
    g0.a[1] = lds_addr;                             // lds_addr (bytes)
    g0.a[2] = (uint32_t)ga;                         // global_addr[31:0]
    g0.a[3] = ((uint32_t)(ga >> 32) & 0x01FFFFFFu)  // global_addr[56:32]
              | 0x80000000u;                        // type=2 ("image") at bits 127:126
    union { uint32_t a[8]; v8u v; } g1;
    g1.a[0] = (2u << 16)      // data_size = 4 bytes
            | (1u << 20)      // pad_enable
            | (3u << 22)      // pad_interval: 16 dwords
            | (3u << 25);     // pad_amount:   4 dwords
    g1.a[1] = (Kd & 0xFFFFu) << 16;                       // tensor_dim0[15:0]
    g1.a[2] = (Kd >> 16) | (128u << 16);                  // tensor_dim0[31:16] | tensor_dim1[15:0]=128
    g1.a[3] = (16u << 16);                                // tile_dim0 = 16 dwords
    g1.a[4] = 128u;                                       // tile_dim1 = 128 rows, tile_dim2 = 0
    g1.a[5] = Kd;                                         // tensor_dim0_stride[31:0] (dwords)
    g1.a[6] = 0u;                                         // stride hi / dim1_stride lo
    g1.a[7] = 0u;
    asm volatile("tensor_load_to_lds %0, %1" :: "s"(g0.v), "s"(g1.v) : "memory");
}

// =====================================================================================
// Transpose + convert: in f32 [R,C] row-major  ->  out bf16 [C,R] row-major
// =====================================================================================
__global__ __launch_bounds__(256) void transpose_f32_bf16(const float* __restrict__ in,
                                                          bf16* __restrict__ out,
                                                          int R, int C) {
    __shared__ float tile[32][33];
    int c0 = blockIdx.x * 32, r0 = blockIdx.y * 32;
    int tx = threadIdx.x & 31, ty = threadIdx.x >> 5;   // 32 x 8
#pragma unroll
    for (int i = 0; i < 32; i += 8) {
        int r = r0 + ty + i, c = c0 + tx;
        if (r < R && c < C) tile[ty + i][tx] = in[(size_t)r * C + c];
    }
    __syncthreads();
#pragma unroll
    for (int i = 0; i < 32; i += 8) {
        int r = r0 + tx, c = c0 + ty + i;
        if (r < R && c < C) out[(size_t)c * R + r] = f2bf(tile[tx][ty + i]);
    }
}

// =====================================================================================
// RMSNorm: f32 [rows, D] -> bf16 [rows, D], one block per row
// =====================================================================================
__global__ __launch_bounds__(256) void rmsnorm_kernel(const float* __restrict__ x,
                                                      const float* __restrict__ w,
                                                      bf16* __restrict__ out, int D) {
    __shared__ float sred[8];
    int row = blockIdx.x;
    const float* xr = x + (size_t)row * D;
    float ss = 0.f;
    for (int d = threadIdx.x; d < D; d += 256) { float v = xr[d]; ss += v * v; }
#pragma unroll
    for (int m = 1; m < 32; m <<= 1) ss += __shfl_xor(ss, m, 32);
    int wv = threadIdx.x >> 5, lane = threadIdx.x & 31;
    if (lane == 0) sred[wv] = ss;
    __syncthreads();
    if (threadIdx.x == 0) {
        float t = 0.f;
#pragma unroll
        for (int i = 0; i < 8; i++) t += sred[i];
        sred[0] = t;
    }
    __syncthreads();
    float scale = rsqrtf(sred[0] / (float)D + 1e-6f);
    for (int d = threadIdx.x; d < D; d += 256)
        out[(size_t)row * D + d] = f2bf(xr[d] * scale * w[d]);
}

// =====================================================================================
// WMMA bf16 GEMM:  C[M,N] = A[M,K] x W[K,N], W pre-transposed as Bt[N,K] (bf16).
// Block = 256 threads (8 waves), tile 128(M) x 128(N), K-step 32.
// Tiles are DMA'd into LDS by the Tensor Data Mover (double buffered); wave 0 programs
// the D# and issues tensor_load_to_lds, all waves compute 8 WMMAs per K-step.
// EPI: 0 -> f32 store, 1 -> bf16 store, 2 -> silu->bf16, 3 -> (*extra)->bf16
// =====================================================================================
#define LDS_STRIDE 20   // 16 data dwords + 4 pad dwords (TDM pad), conflict-free
template <int EPI>
__global__ __launch_bounds__(256) void gemm_bf16_wmma(const bf16* __restrict__ A,
                                                      const bf16* __restrict__ Bt,
                                                      float* __restrict__ Cf,
                                                      bf16* __restrict__ Cb,
                                                      const bf16* __restrict__ extra,
                                                      int M, int N, int K) {
    __shared__ __align__(16) uint32_t a_lds[2][128 * LDS_STRIDE];
    __shared__ __align__(16) uint32_t b_lds[2][128 * LDS_STRIDE];
    const int tid = threadIdx.x;
    const int mBase = blockIdx.y * 128;
    const int nBase = blockIdx.x * 128;
    const int w = tid >> 5, lane = tid & 31;
    const int waveM = w & 3, waveN = w >> 2;      // 4 x 2 waves -> 32x64 per wave
    const int fr = lane & 15;
    const int kh = (lane >> 4) * 4;               // dword K-half offset
    const uint32_t Kd = (uint32_t)(K >> 1);       // dwords per bf16 row

    v8f zero = {0.f, 0.f, 0.f, 0.f, 0.f, 0.f, 0.f, 0.f};
    v8f acc[2][4];
#pragma unroll
    for (int i = 0; i < 2; i++)
#pragma unroll
        for (int j = 0; j < 4; j++) acc[i][j] = zero;

    const uint32_t* gA = (const uint32_t*)A;
    const uint32_t* gB = (const uint32_t*)Bt;
    const int ksteps = K >> 5;

    // prologue: DMA tile 0 into buffer 0
    if (w == 0) {
        tdm_load_tile((uint32_t)(uintptr_t)&a_lds[0][0], gA + (size_t)mBase * Kd, Kd);
        tdm_load_tile((uint32_t)(uintptr_t)&b_lds[0][0], gB + (size_t)nBase * Kd, Kd);
    }

    for (int s_ = 0; s_ < ksteps; s_++) {
        const int cur = s_ & 1;
        if (w == 0) {
            if (s_ + 1 < ksteps) {
                const size_t kdn = (size_t)(s_ + 1) * 16;   // dword column of next tile
                tdm_load_tile((uint32_t)(uintptr_t)&a_lds[cur ^ 1][0],
                              gA + (size_t)mBase * Kd + kdn, Kd);
                tdm_load_tile((uint32_t)(uintptr_t)&b_lds[cur ^ 1][0],
                              gB + (size_t)nBase * Kd + kdn, Kd);
                __builtin_amdgcn_s_wait_tensorcnt(2);   // current pair complete
            } else {
                __builtin_amdgcn_s_wait_tensorcnt(0);
            }
        }
        __syncthreads();   // tile `cur` visible to all waves

        union Frag { uint32_t u[8]; v16bf v; };
        Frag af[2];
#pragma unroll
        for (int mf = 0; mf < 2; mf++) {
            const uint32_t* pa = &a_lds[cur][(waveM * 32 + mf * 16 + fr) * LDS_STRIDE];
#pragma unroll
            for (int i = 0; i < 4; i++) { af[mf].u[i] = pa[kh + i]; af[mf].u[4 + i] = pa[8 + kh + i]; }
        }
#pragma unroll
        for (int nf = 0; nf < 4; nf++) {
            Frag bf_;
            const uint32_t* pb = &b_lds[cur][(waveN * 64 + nf * 16 + fr) * LDS_STRIDE];
#pragma unroll
            for (int i = 0; i < 4; i++) { bf_.u[i] = pb[kh + i]; bf_.u[4 + i] = pb[8 + kh + i]; }
#pragma unroll
            for (int mf = 0; mf < 2; mf++) {
                acc[mf][nf] = __builtin_amdgcn_wmma_f32_16x16x32_bf16(
                    false, af[mf].v, false, bf_.v, (short)0, acc[mf][nf], false, false);
            }
        }
        __syncthreads();   // all waves done reading before buffer reuse
    }

    // ---- epilogue ----
    const int colBase = nBase + waveN * 64 + fr;
#pragma unroll
    for (int mf = 0; mf < 2; mf++) {
        const int rowBase = mBase + waveM * 32 + mf * 16 + ((lane >> 4) << 3);
#pragma unroll
        for (int nf = 0; nf < 4; nf++) {
            int col = colBase + nf * 16;
#pragma unroll
            for (int r = 0; r < 8; r++) {
                int row = rowBase + r;
                float v = acc[mf][nf][r];
                size_t idx = (size_t)row * N + col;
                if (EPI == 0) {
                    Cf[idx] = v;
                } else if (EPI == 1) {
                    Cb[idx] = f2bf(v);
                } else if (EPI == 2) {
                    Cb[idx] = f2bf(v / (1.f + __expf(-v)));   // silu
                } else {
                    Cb[idx] = f2bf(v * bf2f(extra[idx]));     // multiply by gate
                }
            }
        }
    }
}

// =====================================================================================
// Attention prep: from q,k,v,graw (bf16 [BT, H*Dh]) build
//   q2b = q*q, gb = softplus(graw), gkb = g*k (all bf16, same layout),
//   vtb = V transposed per head -> bf16 [B*H, Dh, T],
//   rowc[b*H+h][t] = sum_d g*k*k (f32)
// =====================================================================================
__global__ __launch_bounds__(256) void attn_prep(const bf16* __restrict__ q,
                                                 const bf16* __restrict__ k,
                                                 const bf16* __restrict__ v,
                                                 const bf16* __restrict__ graw,
                                                 bf16* __restrict__ q2b,
                                                 bf16* __restrict__ gb,
                                                 bf16* __restrict__ gkb,
                                                 bf16* __restrict__ vtb,
                                                 float* __restrict__ rowc,
                                                 int T, int H, int Dh) {
    int row = blockIdx.x;            // b*T + t
    int b = row / T, t = row % T;
    int tid = threadIdx.x;
    int head = tid >> 4;             // 16 threads x 4 elems = 64 = Dh
    size_t base = (size_t)row * (H * Dh);
    float s = 0.f;
#pragma unroll
    for (int j = 0; j < 4; j++) {
        int d = tid * 4 + j;
        float qf = bf2f(q[base + d]);
        float kf = bf2f(k[base + d]);
        float vf = bf2f(v[base + d]);
        float gr = bf2f(graw[base + d]);
        float gf = (gr > 20.f) ? gr : log1pf(__expf(gr));   // softplus
        q2b[base + d] = f2bf(qf * qf);
        gb[base + d]  = f2bf(gf);
        float gk = gf * kf;
        gkb[base + d] = f2bf(gk);
        s += gk * kf;
        int dl = d & (Dh - 1);
        vtb[((size_t)(b * H + head) * Dh + dl) * T + t] = f2bf(vf);
    }
#pragma unroll
    for (int m = 1; m < 16; m <<= 1) s += __shfl_xor(s, m, 32);
    if ((tid & 15) == 0) rowc[(size_t)(b * H + head) * T + t] = s;
}

// =====================================================================================
// Flash-style geodesic attention. One wave32 per (b,h, 16-query tile).
//   dist = q2 . g^T - 2 q . (gk)^T + rowc(j);  scores = -dist/8; causal; online softmax;
//   O = P . V via WMMA (P relayout through LDS, V pre-transposed).
// =====================================================================================
__global__ __launch_bounds__(32) void attn_kernel(const bf16* __restrict__ qb,
                                                  const bf16* __restrict__ q2b,
                                                  const bf16* __restrict__ gb,
                                                  const bf16* __restrict__ gkb,
                                                  const bf16* __restrict__ vtb,
                                                  const float* __restrict__ rowc,
                                                  bf16* __restrict__ attb,
                                                  int T, int H, int Dh) {
    __shared__ uint32_t p_lds[16 * 17];   // P tile 16x32 bf16, padded rows
    const int it = blockIdx.x;            // query tile
    const int bh = blockIdx.y;            // b*H + h
    const int b = bh / H, h = bh % H;
    const int lane = threadIdx.x;
    const int fr = lane & 15;
    const int kh = (lane >> 4) * 4;
    const int D = H * Dh;
    const int Dd = D >> 1;

    const uint32_t* qd  = (const uint32_t*)qb;
    const uint32_t* q2d = (const uint32_t*)q2b;
    const uint32_t* gd  = (const uint32_t*)gb;
    const uint32_t* gkd = (const uint32_t*)gkb;
    const uint32_t* vtd = (const uint32_t*)vtb;

    union Frag { uint32_t u[8]; v16bf v; };

    // Q fragments (rows = query tokens), two K-steps of 32 over Dh=64
    Frag qf[2], q2f[2];
    {
        size_t rbase = ((size_t)(b * T + it * 16 + fr)) * Dd + ((h * Dh) >> 1);
#pragma unroll
        for (int ks = 0; ks < 2; ks++) {
            size_t o = rbase + ks * 16;
#pragma unroll
            for (int i = 0; i < 4; i++) {
                qf[ks].u[i]     = qd[o + kh + i];
                qf[ks].u[4 + i] = qd[o + 8 + kh + i];
                q2f[ks].u[i]     = q2d[o + kh + i];
                q2f[ks].u[4 + i] = q2d[o + 8 + kh + i];
            }
        }
    }

    v8f zero = {0.f, 0.f, 0.f, 0.f, 0.f, 0.f, 0.f, 0.f};
    v8f acc[4] = {zero, zero, zero, zero};
    float mrun[8], lrun[8];
#pragma unroll
    for (int r = 0; r < 8; r++) { mrun[r] = -1e30f; lrun[r] = 0.f; }
    const float sc = 0.125f;  // 1/sqrt(Dh)

    for (int jt = 0; jt <= it; jt += 2) {
        float s[2][8];
#pragma unroll
        for (int hf = 0; hf < 2; hf++) {
            int jj = jt + hf;
            if (jj <= it) {
                Frag gf_[2], gkf[2];
                size_t rbase = ((size_t)(b * T + jj * 16 + fr)) * Dd + ((h * Dh) >> 1);
#pragma unroll
                for (int ks = 0; ks < 2; ks++) {
                    size_t o = rbase + ks * 16;
#pragma unroll
                    for (int i = 0; i < 4; i++) {
                        gf_[ks].u[i]     = gd[o + kh + i];
                        gf_[ks].u[4 + i] = gd[o + 8 + kh + i];
                        gkf[ks].u[i]     = gkd[o + kh + i];
                        gkf[ks].u[4 + i] = gkd[o + 8 + kh + i];
                    }
                }
                v8f c1 = zero, c2 = zero;
                c1 = __builtin_amdgcn_wmma_f32_16x16x32_bf16(false, q2f[0].v, false, gf_[0].v, (short)0, c1, false, false);
                c1 = __builtin_amdgcn_wmma_f32_16x16x32_bf16(false, q2f[1].v, false, gf_[1].v, (short)0, c1, false, false);
                c2 = __builtin_amdgcn_wmma_f32_16x16x32_bf16(false, qf[0].v, false, gkf[0].v, (short)0, c2, false, false);
                c2 = __builtin_amdgcn_wmma_f32_16x16x32_bf16(false, qf[1].v, false, gkf[1].v, (short)0, c2, false, false);
                float rc = rowc[(size_t)bh * T + jj * 16 + fr];
                int jcol = jj * 16 + fr;
#pragma unroll
                for (int r = 0; r < 8; r++) {
                    int irow = it * 16 + r + ((lane >> 4) << 3);
                    float dist = c1[r] - 2.f * c2[r] + rc;
                    s[hf][r] = (jcol <= irow) ? (-dist * sc) : -1e30f;
                }
            } else {
#pragma unroll
                for (int r = 0; r < 8; r++) s[hf][r] = -1e30f;
            }
        }
        // ---- online softmax over this 16x32 pair ----
        bf16* pl = (bf16*)p_lds;
#pragma unroll
        for (int r = 0; r < 8; r++) {
            float mx = fmaxf(s[0][r], s[1][r]);
#pragma unroll
            for (int m = 1; m < 16; m <<= 1) mx = fmaxf(mx, __shfl_xor(mx, m, 32));
            float mnew = fmaxf(mrun[r], mx);
            float scale = __expf(mrun[r] - mnew);
            mrun[r] = mnew;
            lrun[r] *= scale;
#pragma unroll
            for (int nt = 0; nt < 4; nt++) acc[nt][r] *= scale;
            float p0 = __expf(s[0][r] - mnew);
            float p1 = __expf(s[1][r] - mnew);
            float rs = p0 + p1;
#pragma unroll
            for (int m = 1; m < 16; m <<= 1) rs += __shfl_xor(rs, m, 32);
            lrun[r] += rs;
            int mr = r + ((lane >> 4) << 3);
            pl[mr * 34 + fr]      = f2bf(p0);
            pl[mr * 34 + 16 + fr] = f2bf(p1);
        }
        __syncthreads();
        // ---- P fragment (A layout) from LDS ----
        Frag pf;
        {
            const uint32_t* pr = &p_lds[fr * 17];
#pragma unroll
            for (int i = 0; i < 4; i++) { pf.u[i] = pr[kh + i]; pf.u[4 + i] = pr[8 + kh + i]; }
        }
        // ---- V fragments (transposed: contiguous along j) + PV WMMA ----
#pragma unroll
        for (int nt = 0; nt < 4; nt++) {
            Frag vf;
            size_t vb = ((size_t)(bh * Dh + nt * 16 + fr)) * (T >> 1) + (size_t)jt * 8;
#pragma unroll
            for (int i = 0; i < 4; i++) { vf.u[i] = vtd[vb + kh + i]; vf.u[4 + i] = vtd[vb + 8 + kh + i]; }
            acc[nt] = __builtin_amdgcn_wmma_f32_16x16x32_bf16(false, pf.v, false, vf.v,
                                                              (short)0, acc[nt], false, false);
        }
        __syncthreads();
    }

    // ---- normalize + store bf16 [BT, D] ----
#pragma unroll
    for (int r = 0; r < 8; r++) {
        float inv = 1.f / lrun[r];
        int mr = r + ((lane >> 4) << 3);
        size_t obase = ((size_t)(b * T + it * 16 + mr)) * D + h * Dh + fr;
#pragma unroll
        for (int nt = 0; nt < 4; nt++)
            attb[obase + nt * 16] = f2bf(acc[nt][r] * inv);
    }
}

// =====================================================================================
// Elementwise residual add (f32)
// =====================================================================================
__global__ __launch_bounds__(256) void add_f32(const float* __restrict__ a,
                                               const float* __restrict__ b,
                                               float* __restrict__ c, size_t n) {
    size_t i = (size_t)blockIdx.x * 256 + threadIdx.x;
    size_t stride = (size_t)gridDim.x * 256;
    for (; i < n; i += stride) c[i] = a[i] + b[i];
}

// =====================================================================================
// Host-side orchestration
// =====================================================================================
extern "C" void kernel_launch(void* const* d_in, const int* in_sizes, int n_in,
                              void* d_out, int out_size, void* d_ws, size_t ws_size,
                              hipStream_t stream) {
    (void)in_sizes; (void)n_in; (void)out_size; (void)ws_size;
    constexpr int B = 4, T = 1024, D = 1024, H = 16, Dh = 64, DF = 4096, MH = 256;
    constexpr int BT = B * T;

    const float* x    = (const float*)d_in[0];
    const float* n1w  = (const float*)d_in[1];
    const float* n2w  = (const float*)d_in[2];
    const float* wq   = (const float*)d_in[3];
    const float* wk   = (const float*)d_in[4];
    const float* wv   = (const float*)d_in[5];
    const float* wo   = (const float*)d_in[6];
    const float* w1   = (const float*)d_in[7];
    const float* w2   = (const float*)d_in[8];
    const float* wg   = (const float*)d_in[9];
    const float* wu   = (const float*)d_in[10];
    const float* wd   = (const float*)d_in[11];
    float* out = (float*)d_out;

    char* ws = (char*)d_ws;
    size_t off = 0;
    auto alloc = [&](size_t bytes) -> char* {
        char* p = ws + off;
        off = (off + bytes + 255) & ~(size_t)255;
        return p;
    };
    bf16* wq_t   = (bf16*)alloc((size_t)D * D * 2);
    bf16* wk_t   = (bf16*)alloc((size_t)D * D * 2);
    bf16* wv_t   = (bf16*)alloc((size_t)D * D * 2);
    bf16* wo_t   = (bf16*)alloc((size_t)D * D * 2);
    bf16* w1_t   = (bf16*)alloc((size_t)MH * D * 2);
    bf16* w2_t   = (bf16*)alloc((size_t)D * MH * 2);
    bf16* wg_t   = (bf16*)alloc((size_t)DF * D * 2);
    bf16* wu_t   = (bf16*)alloc((size_t)DF * D * 2);
    bf16* wd_t   = (bf16*)alloc((size_t)D * DF * 2);
    bf16* h_b    = (bf16*)alloc((size_t)BT * D * 2);
    bf16* qb0    = (bf16*)alloc((size_t)BT * D * 2);
    bf16* kb0    = (bf16*)alloc((size_t)BT * D * 2);
    bf16* vb0    = (bf16*)alloc((size_t)BT * D * 2);
    bf16* m1b    = (bf16*)alloc((size_t)BT * MH * 2);
    bf16* grawb  = (bf16*)alloc((size_t)BT * D * 2);
    bf16* q2b    = (bf16*)alloc((size_t)BT * D * 2);
    bf16* gb     = (bf16*)alloc((size_t)BT * D * 2);
    bf16* gkb    = (bf16*)alloc((size_t)BT * D * 2);
    bf16* vtb    = (bf16*)alloc((size_t)B * H * Dh * T * 2);
    float* rowc  = (float*)alloc((size_t)B * H * T * 4);
    bf16* attb   = (bf16*)alloc((size_t)BT * D * 2);
    float* obuf  = (float*)alloc((size_t)BT * D * 4);   // reused for FFN down output
    float* xmid  = (float*)alloc((size_t)BT * D * 4);
    bf16* h2b    = (bf16*)alloc((size_t)BT * D * 2);
    bf16* gsil   = (bf16*)alloc((size_t)BT * DF * 2);
    bf16* gg     = (bf16*)alloc((size_t)BT * DF * 2);

    // ---- weight transpose/convert to bf16 [N,K] ----
    transpose_f32_bf16<<<dim3(D / 32, D / 32), 256, 0, stream>>>(wq, wq_t, D, D);
    transpose_f32_bf16<<<dim3(D / 32, D / 32), 256, 0, stream>>>(wk, wk_t, D, D);
    transpose_f32_bf16<<<dim3(D / 32, D / 32), 256, 0, stream>>>(wv, wv_t, D, D);
    transpose_f32_bf16<<<dim3(D / 32, D / 32), 256, 0, stream>>>(wo, wo_t, D, D);
    transpose_f32_bf16<<<dim3(MH / 32, D / 32), 256, 0, stream>>>(w1, w1_t, D, MH);
    transpose_f32_bf16<<<dim3(D / 32, MH / 32), 256, 0, stream>>>(w2, w2_t, MH, D);
    transpose_f32_bf16<<<dim3(DF / 32, D / 32), 256, 0, stream>>>(wg, wg_t, D, DF);
    transpose_f32_bf16<<<dim3(DF / 32, D / 32), 256, 0, stream>>>(wu, wu_t, D, DF);
    transpose_f32_bf16<<<dim3(D / 32, DF / 32), 256, 0, stream>>>(wd, wd_t, DF, D);

    // ---- pre-norm 1 ----
    rmsnorm_kernel<<<BT, 256, 0, stream>>>(x, n1w, h_b, D);

    // ---- Q/K/V + MetricNet GEMMs (bf16 WMMA, TDM-staged) ----
    dim3 gD(D / 128, BT / 128);
    gemm_bf16_wmma<1><<<gD, 256, 0, stream>>>(h_b, wq_t, nullptr, qb0, nullptr, BT, D, D);
    gemm_bf16_wmma<1><<<gD, 256, 0, stream>>>(h_b, wk_t, nullptr, kb0, nullptr, BT, D, D);
    gemm_bf16_wmma<1><<<gD, 256, 0, stream>>>(h_b, wv_t, nullptr, vb0, nullptr, BT, D, D);
    gemm_bf16_wmma<2><<<dim3(MH / 128, BT / 128), 256, 0, stream>>>(h_b, w1_t, nullptr, m1b, nullptr, BT, MH, D);
    gemm_bf16_wmma<1><<<gD, 256, 0, stream>>>(m1b, w2_t, nullptr, grawb, nullptr, BT, D, MH);

    // ---- attention prep + flash attention ----
    attn_prep<<<BT, 256, 0, stream>>>(qb0, kb0, vb0, grawb, q2b, gb, gkb, vtb, rowc, T, H, Dh);
    attn_kernel<<<dim3(T / 16, B * H), 32, 0, stream>>>(qb0, q2b, gb, gkb, vtb, rowc, attb, T, H, Dh);

    // ---- output projection + residual ----
    gemm_bf16_wmma<0><<<gD, 256, 0, stream>>>(attb, wo_t, obuf, nullptr, nullptr, BT, D, D);
    add_f32<<<4096, 256, 0, stream>>>(x, obuf, xmid, (size_t)BT * D);

    // ---- pre-norm 2 + SwiGLU FFN ----
    rmsnorm_kernel<<<BT, 256, 0, stream>>>(xmid, n2w, h2b, D);
    dim3 gF(DF / 128, BT / 128);
    gemm_bf16_wmma<2><<<gF, 256, 0, stream>>>(h2b, wg_t, nullptr, gsil, nullptr, BT, DF, D);
    gemm_bf16_wmma<3><<<gF, 256, 0, stream>>>(h2b, wu_t, nullptr, gg, gsil, BT, DF, D);
    gemm_bf16_wmma<0><<<gD, 256, 0, stream>>>(gg, wd_t, obuf, nullptr, nullptr, BT, D, DF);
    add_f32<<<4096, 256, 0, stream>>>(xmid, obuf, out, (size_t)BT * D);
}
